// GCN_v2_38620345926217
// MI455X (gfx1250) — compile-verified, matrix operand order checked by reading, not run on hardware
//
#include <hip/hip_runtime.h>
#include <hip/hip_bf16.h>

#define NEG_SLOPE 0.2f

typedef __attribute__((ext_vector_type(16))) _Float16 v16h;
typedef __attribute__((ext_vector_type(8)))  float    v8f;

// ---------------------------------------------------------------------------
// Node transform:  Out[n, CO] = X[n, 64] @ W[64, CO] + bias   (f16 WMMA, f32 acc)
// One wave (32 threads) computes a 16x16 output tile; K=64 -> two 16x16x32 WMMAs.
// A fragment (16x32 f16): lane L holds row M=L%16; half-wave group g=L/16 owns
//   K offsets {g*8 .. g*8+7} within each 16-wide K half (per ISA 7.12.2 table).
// B fragment (32x16 f16): lane L holds column N=L%16 with the same K striping.
// C/D (16x16 f32): VGPR r, lane L -> (M = r + 8*(L/16), N = L%16).
// ---------------------------------------------------------------------------
template<int CO>
__global__ __launch_bounds__(32)
void node_xform_wmma(const float* __restrict__ X, const float* __restrict__ W,
                     const float* __restrict__ bias, float* __restrict__ Out,
                     int n_nodes) {
  const int mtile = blockIdx.x;
  const int ntile = blockIdx.y;
  const int lane  = threadIdx.x;          // 0..31 (wave32)
  const int mrow  = lane & 15;
  const int grp   = lane >> 4;            // 0 or 1
  const int nIdx  = ntile * 16 + (lane & 15);

  int node = mtile * 16 + mrow;
  if (node >= n_nodes) node = n_nodes - 1;   // branchless read clamp

  const float bv = bias[nIdx];
  v8f c;
#pragma unroll
  for (int r = 0; r < 8; ++r) c[r] = bv;     // bias folded into accumulator init

#pragma unroll
  for (int ks = 0; ks < 2; ++ks) {
    v16h a, b;
#pragma unroll
    for (int v = 0; v < 8; ++v) {
      const int k = ks * 32 + ((v & 4) ? 16 : 0) + grp * 8 + (v & 3) * 2;
      const float2 xp = *(const float2*)(X + (size_t)node * 64 + k);
      a[2 * v]     = (_Float16)xp.x;
      a[2 * v + 1] = (_Float16)xp.y;
      b[2 * v]     = (_Float16)W[(size_t)(k)     * CO + nIdx];
      b[2 * v + 1] = (_Float16)W[(size_t)(k + 1) * CO + nIdx];
    }
    // (neg_a, A, neg_b, B, c_mod, C, reuse_a, reuse_b)
    c = __builtin_amdgcn_wmma_f32_16x16x32_f16(false, a, false, b,
                                               (short)0, c, false, false);
  }

  // Tile-uniform bounds check: full tiles take the scalar-branch fast path
  // with 8 unguarded coalesced stores (no per-store EXEC churn).
  if ((mtile + 1) * 16 <= n_nodes) {
#pragma unroll
    for (int r = 0; r < 8; ++r)
      Out[(size_t)(mtile * 16 + r + 8 * grp) * CO + nIdx] = c[r];
  } else {
#pragma unroll
    for (int r = 0; r < 8; ++r) {
      const int row = mtile * 16 + r + 8 * grp;
      if (row < n_nodes) Out[(size_t)row * CO + nIdx] = c[r];
    }
  }
}

// ---------------------------------------------------------------------------
// Per-layer init: zero the scatter accumulator, set amax=-FLT_MAX, den=0.
// ---------------------------------------------------------------------------
__global__ void k_init(float* __restrict__ amax, float* __restrict__ den,
                       float* __restrict__ acc, int n, int co) {
  const int i = blockIdx.x * blockDim.x + threadIdx.x;
  if (i < n * co) acc[i] = 0.0f;
  if (i < n) { amax[i] = -3.402823466e38f; den[i] = 0.0f; }
}

// Ordered-int trick for float atomic max (amax initialized to -FLT_MAX).
__device__ __forceinline__ void atomicMaxFloat(float* addr, float val) {
  if (val >= 0.0f) atomicMax((int*)addr, __float_as_int(val));
  else             atomicMin((unsigned int*)addr, __float_as_uint(val));
}

// ---------------------------------------------------------------------------
// Edge pass 1: alpha[e] = att . leaky_relu(xl[src] + xr[dst] + ef*We)
// CO/4 lanes per edge, float4 (b128) gathers, shfl reduction, atomic max.
// ---------------------------------------------------------------------------
template<int CO>
__global__ __launch_bounds__(256)
void edge_alpha(const float* __restrict__ xl, const float* __restrict__ xr,
                const int* __restrict__ src, const int* __restrict__ dst,
                const float* __restrict__ ef, const float* __restrict__ We,
                const float* __restrict__ att,
                float* __restrict__ alpha, float* __restrict__ amax, int ne) {
  constexpr int LPE = CO / 4;
  const int gid = blockIdx.x * blockDim.x + threadIdx.x;
  const int e   = gid / LPE;
  const int sub = gid % LPE;
  if (e >= ne) return;                       // uniform within each LPE subgroup
  const int s = src[e], d = dst[e];
  const float fe = ef[e];
  const int off = sub * 4;
  const float4 a4 = *(const float4*)(xl + (size_t)s * CO + off);
  const float4 r4 = *(const float4*)(xr + (size_t)d * CO + off);
  const float4 w4 = *(const float4*)(We + off);
  const float4 t4 = *(const float4*)(att + off);
  float acc = 0.0f, m;
  m = a4.x + r4.x + fe * w4.x; m = (m > 0.0f) ? m : NEG_SLOPE * m; acc += m * t4.x;
  m = a4.y + r4.y + fe * w4.y; m = (m > 0.0f) ? m : NEG_SLOPE * m; acc += m * t4.y;
  m = a4.z + r4.z + fe * w4.z; m = (m > 0.0f) ? m : NEG_SLOPE * m; acc += m * t4.z;
  m = a4.w + r4.w + fe * w4.w; m = (m > 0.0f) ? m : NEG_SLOPE * m; acc += m * t4.w;
#pragma unroll
  for (int o = LPE / 2; o > 0; o >>= 1) acc += __shfl_down(acc, o, LPE);
  if (sub == 0) { alpha[e] = acc; atomicMaxFloat(amax + d, acc); }
}

// ---------------------------------------------------------------------------
// Edge pass 2: ex = exp(alpha - amax[dst]); den[dst] += ex (in-place on alpha)
// ---------------------------------------------------------------------------
__global__ __launch_bounds__(256)
void edge_exp(float* __restrict__ alpha, const int* __restrict__ dst,
              const float* __restrict__ amax, float* __restrict__ den, int ne) {
  const int e = blockIdx.x * blockDim.x + threadIdx.x;
  if (e >= ne) return;
  const int d = dst[e];
  const float ex = __expf(alpha[e] - amax[d]);
  alpha[e] = ex;
  atomicAdd(den + d, ex);
}

// ---------------------------------------------------------------------------
// Edge pass 3: acc[dst] += xl[src] * (ex / (den[dst] + 1e-16))
// ---------------------------------------------------------------------------
template<int CO>
__global__ __launch_bounds__(256)
void edge_scatter(const float* __restrict__ xl, const float* __restrict__ ex,
                  const float* __restrict__ den, const int* __restrict__ src,
                  const int* __restrict__ dst, float* __restrict__ acc, int ne) {
  constexpr int LPE = CO / 4;
  const int gid = blockIdx.x * blockDim.x + threadIdx.x;
  const int e   = gid / LPE;
  const int sub = gid % LPE;
  if (e >= ne) return;
  const int s = src[e], d = dst[e];
  const float w = ex[e] / (den[d] + 1e-16f);
  const int off = sub * 4;
  const float4 v = *(const float4*)(xl + (size_t)s * CO + off);
  float* p = acc + (size_t)d * CO + off;
  atomicAdd(p + 0, v.x * w);
  atomicAdd(p + 1, v.y * w);
  atomicAdd(p + 2, v.z * w);
  atomicAdd(p + 3, v.w * w);
}

// ---------------------------------------------------------------------------
// Finalize: in-place bias add (+ optional ReLU)
// ---------------------------------------------------------------------------
__global__ __launch_bounds__(256)
void k_finalize(float* __restrict__ acc, const float* __restrict__ bias,
                int n, int co, int do_relu) {
  const int i = blockIdx.x * blockDim.x + threadIdx.x;
  if (i >= n * co) return;
  float v = acc[i] + bias[i % co];   // co is a power of two -> AND
  if (do_relu) v = fmaxf(v, 0.0f);
  acc[i] = v;
}

// ---------------------------------------------------------------------------
extern "C" void kernel_launch(void* const* d_in, const int* in_sizes, int n_in,
                              void* d_out, int out_size, void* d_ws, size_t ws_size,
                              hipStream_t stream) {
  (void)n_in; (void)out_size; (void)ws_size;
  const float* x    = (const float*)d_in[0];
  const int*   ei   = (const int*)d_in[1];
  const float* ef   = (const float*)d_in[2];
  const float* Wl0  = (const float*)d_in[3];
  const float* bl0  = (const float*)d_in[4];
  const float* Wr0  = (const float*)d_in[5];
  const float* br0  = (const float*)d_in[6];
  const float* We0  = (const float*)d_in[7];
  const float* att0 = (const float*)d_in[8];
  const float* b0   = (const float*)d_in[9];
  const float* Wl1  = (const float*)d_in[10];
  const float* bl1  = (const float*)d_in[11];
  const float* Wr1  = (const float*)d_in[12];
  const float* br1  = (const float*)d_in[13];
  const float* We1  = (const float*)d_in[14];
  const float* att1 = (const float*)d_in[15];
  const float* b1   = (const float*)d_in[16];

  const int n = in_sizes[0] / 64;   // 50000 nodes
  const int e = in_sizes[1] / 2;    // 800000 edges
  const int* srcI = ei;
  const int* dstI = ei + e;

  // Workspace carve-up (all f32): ~42 MB, L2-resident on MI455X.
  float* ws   = (float*)d_ws;
  float* xl   = ws;  ws += (size_t)n * 64;
  float* xr   = ws;  ws += (size_t)n * 64;
  float* hbuf = ws;  ws += (size_t)n * 64;
  float* alp  = ws;  ws += (size_t)e;
  float* amax = ws;  ws += n;
  float* den  = ws;  ws += n;

  auto cdiv = [](long long a, long long b) { return (int)((a + b - 1) / b); };
  const int B = 256;

  // ------------------------- Layer 0 (CO = 64) -------------------------
  {
    constexpr int CO = 64;
    k_init<<<cdiv((long long)n * CO, B), B, 0, stream>>>(amax, den, hbuf, n, CO);
    dim3 g((n + 15) / 16, CO / 16);
    node_xform_wmma<CO><<<g, 32, 0, stream>>>(x, Wl0, bl0, xl, n);
    node_xform_wmma<CO><<<g, 32, 0, stream>>>(x, Wr0, br0, xr, n);
    edge_alpha<CO><<<cdiv((long long)e * (CO / 4), B), B, 0, stream>>>(
        xl, xr, srcI, dstI, ef, We0, att0, alp, amax, e);
    edge_exp<<<cdiv(e, B), B, 0, stream>>>(alp, dstI, amax, den, e);
    edge_scatter<CO><<<cdiv((long long)e * (CO / 4), B), B, 0, stream>>>(
        xl, alp, den, srcI, dstI, hbuf, e);
    k_finalize<<<cdiv((long long)n * CO, B), B, 0, stream>>>(hbuf, b0, n, CO, 1);
  }

  // ------------------------- Layer 1 (CO = 32) -------------------------
  {
    constexpr int CO = 32;
    float* outp = (float*)d_out;
    k_init<<<cdiv((long long)n * CO, B), B, 0, stream>>>(amax, den, outp, n, CO);
    dim3 g((n + 15) / 16, CO / 16);
    node_xform_wmma<CO><<<g, 32, 0, stream>>>(hbuf, Wl1, bl1, xl, n);
    node_xform_wmma<CO><<<g, 32, 0, stream>>>(hbuf, Wr1, br1, xr, n);
    edge_alpha<CO><<<cdiv((long long)e * (CO / 4), B), B, 0, stream>>>(
        xl, xr, srcI, dstI, ef, We1, att1, alp, amax, e);
    edge_exp<<<cdiv(e, B), B, 0, stream>>>(alp, dstI, amax, den, e);
    edge_scatter<CO><<<cdiv((long long)e * (CO / 4), B), B, 0, stream>>>(
        xl, alp, den, srcI, dstI, outp, e);
    k_finalize<<<cdiv((long long)n * CO, B), B, 0, stream>>>(outp, b1, n, CO, 0);
  }
}
